// MVSNet_43490838839954
// MI455X (gfx1250) — matrix operand order repaired and checked by reading, not compiled, not run on hardware
//
#include <hip/hip_runtime.h>
#include <hip/hip_bf16.h>
#include <math.h>

// ===========================================================================
// MVSNet forward for MI455X (gfx1250, wave32, WMMA).
//
// gru_cell is always called with h == 0  =>  r-conv is dead and wu/wc only use
// their first cin input slices:
//   h_out = (1 - sigmoid(conv(x, wu[:, :cin]) + bu)) * tanh(conv(x, wc[:, :cin]) + bc)
// GRU1 => M=16, K=288(=9x32), N=20480 implicit GEMM per depth slice, mapped to
// v_wmma_f32_16x16x32_f16 (f16 in, f32 accumulate). Tap-major K ordering + HWC
// f16 activations make each lane's B fragment a contiguous 32-byte load for
// cin=32/16. Chunk loops are kept ROLLED to avoid register spills (round-2 asm
// showed ScratchSize=1828 with full unroll).
//
// Input flattening (setup_inputs() insertion order, depth-first):
//   0 imgs(1,3,3,512,640)  1 proj(1,3,4,4)  2 depth_values(1,48)
//   3..7   c0 {w,gamma,beta,mean,var}   8..12 c1   13..17 c2   18..22 c3
//   23..27 c4   28..32 c5   33..37 c6   38 feat_w  39 feat_b
//   40..45 g1 {wr,br,wu,bu,wc,bc}   46..51 g2   52..57 g3
//   58 prob_w  59 prob_b
// ===========================================================================

typedef __attribute__((ext_vector_type(16))) _Float16 v16h;
typedef __attribute__((ext_vector_type(8)))  float    v8f;

#define IMG_H 512
#define IMG_W 640
#define FH    128
#define FW    160
#define FC    32
#define ND    48
#define NPIX  (FH*FW)

// --------------------------------------------------------------------------
// Feature-net conv (fp32, NCHW), templated on CIN and kernel size K. Taps
// fully unrolled; ic-loop unroll bounded (4) to limit register pressure.
// mode 0: BN+ReLU, mode 1: bias only.
// --------------------------------------------------------------------------
template<int CIN, int KK>
__global__ void conv_bn_relu_t(const float* __restrict__ in, float* __restrict__ out,
                               const float* __restrict__ w,
                               const float* __restrict__ gamma, const float* __restrict__ beta,
                               const float* __restrict__ mean,  const float* __restrict__ var,
                               const float* __restrict__ bias,
                               int cout, int inH, int inW, int outH, int outW,
                               int stride, int mode)
{
    constexpr int PAD = KK / 2;
    int t = blockIdx.x * blockDim.x + threadIdx.x;
    int total = cout * outH * outW;
    if (t >= total) return;
    int ox = t % outW;
    int tmp = t / outW;
    int oy = tmp % outH;
    int oc = tmp / outH;

    float acc = 0.0f;
    #pragma unroll 4
    for (int ic = 0; ic < CIN; ++ic) {
        const float* ip = in + (size_t)ic * inH * inW;
        const float* wp = w + (((size_t)oc * CIN + ic) * KK) * KK;
        #pragma unroll
        for (int kh = 0; kh < KK; ++kh) {
            int iy = oy * stride + kh - PAD;
            bool vy = (iy >= 0) && (iy < inH);
            #pragma unroll
            for (int kw = 0; kw < KK; ++kw) {
                int ix = ox * stride + kw - PAD;
                bool v = vy && (ix >= 0) && (ix < inW);
                float xv = v ? ip[iy * inW + ix] : 0.0f;
                acc = fmaf(xv, wp[kh * KK + kw], acc);
            }
        }
    }
    if (mode == 0) {
        float s = gamma[oc] * rsqrtf(var[oc] + 1e-5f);
        acc = (acc - mean[oc]) * s + beta[oc];
        acc = fmaxf(acc, 0.0f);
    } else {
        acc += bias[oc];
    }
    out[t] = acc;
}

// --------------------------------------------------------------------------
// One-thread: inv(ref_proj) via Gauss-Jordan, then M_v = P_v * inv(P_0).
// Ms layout per view: 9 rot (row-major) + 3 trans = 12 floats.
// --------------------------------------------------------------------------
__global__ void setup_proj_kernel(const float* __restrict__ P, float* __restrict__ Ms)
{
    if (threadIdx.x != 0 || blockIdx.x != 0) return;
    float A[4][8];
    for (int i = 0; i < 4; ++i)
        for (int j = 0; j < 4; ++j) {
            A[i][j] = P[i * 4 + j];
            A[i][4 + j] = (i == j) ? 1.0f : 0.0f;
        }
    for (int col = 0; col < 4; ++col) {
        int piv = col;
        float best = fabsf(A[col][col]);
        for (int r = col + 1; r < 4; ++r) {
            float v = fabsf(A[r][col]);
            if (v > best) { best = v; piv = r; }
        }
        if (piv != col)
            for (int j = 0; j < 8; ++j) { float tt = A[col][j]; A[col][j] = A[piv][j]; A[piv][j] = tt; }
        float invp = 1.0f / A[col][col];
        for (int j = 0; j < 8; ++j) A[col][j] *= invp;
        for (int r = 0; r < 4; ++r) {
            if (r == col) continue;
            float f = A[r][col];
            for (int j = 0; j < 8; ++j) A[r][j] -= f * A[col][j];
        }
    }
    for (int v = 1; v < 3; ++v) {
        const float* Pv = P + v * 16;
        float* o = Ms + (v - 1) * 12;
        for (int i = 0; i < 3; ++i) {
            for (int j = 0; j < 4; ++j) {
                float s = 0.0f;
                for (int kk = 0; kk < 4; ++kk) s += Pv[i * 4 + kk] * A[kk][4 + j];
                if (j < 3) o[i * 3 + j] = s; else o[9 + i] = s;
            }
        }
    }
}

// --------------------------------------------------------------------------
// Bilinear sample (per-corner validity, clamp gather) — matches reference.
// --------------------------------------------------------------------------
__device__ __forceinline__ float bsample(const float* __restrict__ fc, int H, int W,
                                         float gx, float gy)
{
    float x0 = floorf(gx), y0 = floorf(gy);
    float x1 = x0 + 1.0f, y1 = y0 + 1.0f;
    float wx1 = gx - x0, wx0 = 1.0f - wx1;
    float wy1 = gy - y0, wy0 = 1.0f - wy1;
    float acc = 0.0f;
    #pragma unroll
    for (int q = 0; q < 4; ++q) {
        float xi = (q & 1) ? x1 : x0;
        float yi = (q & 2) ? y1 : y0;
        float wq = ((q & 1) ? wx1 : wx0) * ((q & 2) ? wy1 : wy0);
        bool valid = (xi >= 0.0f) && (xi <= (float)(W - 1)) && (yi >= 0.0f) && (yi <= (float)(H - 1));
        int xc = (int)fminf(fmaxf(xi, 0.0f), (float)(W - 1));
        int yc = (int)fminf(fmaxf(yi, 0.0f), (float)(H - 1));
        float vv = fc[yc * W + xc];
        acc = fmaf(valid ? vv : 0.0f, wq, acc);
    }
    return acc;
}

// --------------------------------------------------------------------------
// Per-depth: warp both src views, variance-cost, negate, store HWC f16 slice.
// V == 3 views (fixed by setup_inputs).
// --------------------------------------------------------------------------
__global__ void warp_cost_kernel(const float* __restrict__ reff,
                                 const float* __restrict__ src0,
                                 const float* __restrict__ src1,
                                 const float* __restrict__ Ms,
                                 const float* __restrict__ dvals, int d,
                                 _Float16* __restrict__ xout)
{
    int t = blockIdx.x * blockDim.x + threadIdx.x;
    if (t >= NPIX * FC) return;
    int c = t / NPIX;
    int pix = t - c * NPIX;
    float fx = (float)(pix % FW);
    float fy = (float)(pix / FW);
    float depth = dvals[d];

    float s1 = 0.0f, s2 = 0.0f;
    #pragma unroll
    for (int v = 0; v < 2; ++v) {
        const float* M = Ms + v * 12;
        float rx = M[0] * fx + M[1] * fy + M[2];
        float ry = M[3] * fx + M[4] * fy + M[5];
        float rz = M[6] * fx + M[7] * fy + M[8];
        float px = rx * depth + M[9];
        float py = ry * depth + M[10];
        float pz = rz * depth + M[11];
        float gx = px / pz, gy = py / pz;
        const float* src = (v == 0) ? src0 : src1;
        float wv = bsample(src + (size_t)c * NPIX, FH, FW, gx, gy);
        s1 += wv;
        s2 = fmaf(wv, wv, s2);
    }
    float ref = reff[(size_t)c * NPIX + pix];
    const float invV = 1.0f / 3.0f;
    float ave  = (ref + s1) * invV;
    float ave2 = (fmaf(ref, ref, s2)) * invV;
    float xv = -(ave2 - ave * ave);
    xout[(size_t)pix * FC + c] = (_Float16)xv;   // HWC for WMMA B-fragment loads
}

// --------------------------------------------------------------------------
// Pack conv weights (OIHW f32, only first `cin` input slices of `cat`) into
// a WMMA A-fragment layout (16-bit A 16x32) in contiguous k order, same as B (not the ISA table order):
//   lane m = lane&15 holds row m; element e -> K = e + 16*(lane>>4)
// K linearized tap-major: k = (kh*3+kw)*cin + c; zero-padded to 32*nchunk.
// Output: wpk[(chunk*32 + lane)*16 + e] halves.
// --------------------------------------------------------------------------
__global__ void pack_w_kernel(const float* __restrict__ w, _Float16* __restrict__ wpk,
                              int cat, int cin, int cout, int nchunk)
{
    int t = blockIdx.x * blockDim.x + threadIdx.x;
    int total = nchunk * 32 * 16;
    if (t >= total) return;
    int e = t & 15;
    int lane = (t >> 4) & 31;
    int chunk = t >> 9;
    int half = lane >> 4;
    int m = lane & 15;
    int k = chunk * 32 + e + 16 * half;
    int K = cin * 9;
    float v = 0.0f;
    if (m < cout && k < K) {
        int tap = k / cin;
        int c = k - tap * cin;
        int kh = tap / 3, kw = tap - kh * 3;
        v = w[(((size_t)m * cat + c) * 3 + kh) * 3 + kw];
    }
    wpk[t] = (_Float16)v;
}

// --------------------------------------------------------------------------
// Gated-conv (degenerate GRU) via WMMA. Chunk loop ROLLED (unroll 1): the
// loop body is one masked 32B B load + two 32B A loads + two WMMAs, so the
// live set stays tiny (no spills). Tap math at runtime is trivial: for
// CIN=32, tap == ch; for CIN=16, tap = 2*ch + half; tap/3 is a constant
// divide (magic multiply).
// One wave = 16(cout) x 16(pixels) tile for BOTH u-conv and c-conv (shared B).
// x: HWC f16 (CIN ch). out: HWC f16 (COUT ch).
// B fragment (16-bit B 32x16, contiguous k order, same as A packing): lane n = lane&15
// holds column n; element e -> K = e + 16*(lane>>4).
// D fragment: row = r + 8*(lane>>4), col = lane&15.
// --------------------------------------------------------------------------
template<int CIN, int COUT, int NCHUNK>
__global__ void gru_conv_wmma_t(const _Float16* __restrict__ xin,
                                const _Float16* __restrict__ wu_pk,
                                const _Float16* __restrict__ wc_pk,
                                const float* __restrict__ bu,
                                const float* __restrict__ bc,
                                _Float16* __restrict__ hout)
{
    constexpr int K = CIN * 9;
    const int lane = threadIdx.x & 31;
    const int wave = threadIdx.x >> 5;
    const int tile = blockIdx.x * (blockDim.x >> 5) + wave;
    const int ntile = NPIX >> 4;            // 20480/16 = 1280, exact
    if (tile >= ntile) return;              // wave-uniform: EXEC stays all-ones
    const int half = lane >> 4;
    const int l15 = lane & 15;

    const int pix = tile * 16 + l15;        // B column = this pixel
    const int py = pix / FW;
    const int px = pix - py * FW;

    v8f accu = {};
    v8f accc = {};
    #pragma unroll 1
    for (int ch = 0; ch < NCHUNK; ++ch) {
        v16h b;
        if constexpr (CIN == 32 || CIN == 16) {
            // 16 consecutive K values = 16 consecutive channels at one tap.
            int kbase = ch * 32 + half * 16;
            int tap = kbase / CIN;          // CIN=32: == ch; CIN=16: 2ch+half
            int c0  = kbase - tap * CIN;
            int kh = tap / 3, kw = tap - kh * 3;
            int yy = py + kh - 1, xx = px + kw - 1;
            v16h tmp = {};
            if (kbase < K && yy >= 0 && yy < FH && xx >= 0 && xx < FW)
                tmp = *(const v16h*)(xin + ((size_t)(yy * FW + xx)) * CIN + c0);
            b = tmp;
        } else {
            #pragma unroll
            for (int e = 0; e < 16; ++e) {
                int k = ch * 32 + half * 16 + e;
                float v = 0.0f;
                if (k < K) {
                    int tap = k / CIN;      // CIN pow2 -> shifts
                    int c = k - tap * CIN;
                    int kh = tap / 3, kw = tap - kh * 3;
                    int yy = py + kh - 1, xx = px + kw - 1;
                    if (yy >= 0 && yy < FH && xx >= 0 && xx < FW)
                        v = (float)xin[((size_t)yy * FW + xx) * CIN + c];
                }
                b[e] = (_Float16)v;
            }
        }
        v16h au = *(const v16h*)(wu_pk + ((size_t)ch * 32 + lane) * 16);
        v16h ac = *(const v16h*)(wc_pk + ((size_t)ch * 32 + lane) * 16);
        accu = __builtin_amdgcn_wmma_f32_16x16x32_f16(false, au, false, b,
                                                      (short)0, accu, false, false);
        accc = __builtin_amdgcn_wmma_f32_16x16x32_f16(false, ac, false, b,
                                                      (short)0, accc, false, false);
    }
    #pragma unroll
    for (int r = 0; r < 8; ++r) {
        int m = r + 8 * half;               // cout row
        if (m < COUT) {
            float u = accu[r] + bu[m];
            float cv = accc[r] + bc[m];
            u = 1.0f / (1.0f + expf(-u));
            cv = tanhf(cv);
            hout[(size_t)pix * COUT + m] = (_Float16)((1.0f - u) * cv);
        }
    }
}

// --------------------------------------------------------------------------
// prob conv: (1,2,3,3) over h3 (HWC f16, 2ch) -> reg[d][pix] (f32).
// --------------------------------------------------------------------------
__global__ void prob_conv_kernel(const _Float16* __restrict__ h3,
                                 const float* __restrict__ pw,
                                 const float* __restrict__ pb,
                                 float* __restrict__ reg, int d)
{
    int pix = blockIdx.x * blockDim.x + threadIdx.x;
    if (pix >= NPIX) return;
    int py = pix / FW, px = pix - py * FW;
    float acc = pb[0];
    #pragma unroll
    for (int kh = 0; kh < 3; ++kh) {
        int yy = py + kh - 1;
        bool vy = (yy >= 0) && (yy < FH);
        #pragma unroll
        for (int kw = 0; kw < 3; ++kw) {
            int xx = px + kw - 1;
            if (vy && xx >= 0 && xx < FW) {
                const _Float16* hp = h3 + ((size_t)yy * FW + xx) * 2;
                acc = fmaf((float)hp[0], pw[(0 * 3 + kh) * 3 + kw], acc);
                acc = fmaf((float)hp[1], pw[(1 * 3 + kh) * 3 + kw], acc);
            }
        }
    }
    reg[(size_t)d * NPIX + pix] = acc;
}

// --------------------------------------------------------------------------
// Softmax over D=48 depths per pixel -> d_out (1,48,128,160).
// --------------------------------------------------------------------------
__global__ void softmax_kernel(const float* __restrict__ reg, float* __restrict__ out)
{
    int pix = blockIdx.x * blockDim.x + threadIdx.x;
    if (pix >= NPIX) return;
    float m = -1e30f;
    for (int d = 0; d < ND; ++d) m = fmaxf(m, reg[(size_t)d * NPIX + pix]);
    float s = 0.0f;
    for (int d = 0; d < ND; ++d) s += expf(reg[(size_t)d * NPIX + pix] - m);
    float inv = 1.0f / s;
    for (int d = 0; d < ND; ++d)
        out[(size_t)d * NPIX + pix] = expf(reg[(size_t)d * NPIX + pix] - m) * inv;
}

// ===========================================================================
// Host orchestration
// ===========================================================================
extern "C" void kernel_launch(void* const* d_in, const int* in_sizes, int n_in,
                              void* d_out, int out_size, void* d_ws, size_t ws_size,
                              hipStream_t stream)
{
    const float* imgs  = (const float*)d_in[0];
    const float* projm = (const float*)d_in[1];
    const float* dvals = (const float*)d_in[2];
    auto P = [&](int i) -> const float* { return (const float*)d_in[i]; };

    // ---- workspace carve (~35 MB) ----
    char* base = (char*)d_ws;
    size_t off = 0;
    auto carve = [&](size_t bytes) -> char* {
        char* p = base + off;
        off = (off + bytes + 255) & ~(size_t)255;
        return p;
    };
    float*    Ms    = (float*)carve(24 * sizeof(float));
    float*    buf0  = (float*)carve((size_t)8 * IMG_H * IMG_W * sizeof(float));
    float*    buf1  = (float*)carve((size_t)8 * IMG_H * IMG_W * sizeof(float));
    float*    feats = (float*)carve((size_t)3 * FC * NPIX * sizeof(float));
    _Float16* xsl   = (_Float16*)carve((size_t)NPIX * FC * sizeof(_Float16));
    _Float16* h1    = (_Float16*)carve((size_t)NPIX * 16 * sizeof(_Float16));
    _Float16* h2    = (_Float16*)carve((size_t)NPIX * 4  * sizeof(_Float16));
    _Float16* h3    = (_Float16*)carve((size_t)NPIX * 2  * sizeof(_Float16));
    float*    regb  = (float*)carve((size_t)ND * NPIX * sizeof(float));
    const int nch1 = 9, nch2 = 5, nch3 = 2;   // ceil(cin*9/32) for cin=32,16,4
    _Float16* g1u = (_Float16*)carve((size_t)nch1 * 512 * sizeof(_Float16));
    _Float16* g1c = (_Float16*)carve((size_t)nch1 * 512 * sizeof(_Float16));
    _Float16* g2u = (_Float16*)carve((size_t)nch2 * 512 * sizeof(_Float16));
    _Float16* g2c = (_Float16*)carve((size_t)nch2 * 512 * sizeof(_Float16));
    _Float16* g3u = (_Float16*)carve((size_t)nch3 * 512 * sizeof(_Float16));
    _Float16* g3c = (_Float16*)carve((size_t)nch3 * 512 * sizeof(_Float16));

    // ---- projection setup ----
    setup_proj_kernel<<<1, 32, 0, stream>>>(projm, Ms);

    // ---- pack GRU weights into WMMA A-fragment layout (u & c; r is dead) ----
    auto pack = [&](const float* w, _Float16* dst, int cat, int cin, int cout, int nch) {
        int total = nch * 512;
        pack_w_kernel<<<(total + 255) / 256, 256, 0, stream>>>(w, dst, cat, cin, cout, nch);
    };
    pack(P(42), g1u, 48, 32, 16, nch1); pack(P(44), g1c, 48, 32, 16, nch1);
    pack(P(48), g2u, 20, 16, 4,  nch2); pack(P(50), g2c, 20, 16, 4,  nch2);
    pack(P(54), g3u, 6,  4,  2,  nch3); pack(P(56), g3c, 6,  4,  2,  nch3);

    // ---- feature net on 3 views ----
    for (int v = 0; v < 3; ++v) {
        const float* img = imgs + (size_t)v * 3 * IMG_H * IMG_W;
        float* fv = feats + (size_t)v * FC * NPIX;
        int g;
        // c0: 3->8, k3, 512x640
        g = (8 * 512 * 640 + 255) / 256;
        conv_bn_relu_t<3, 3><<<g, 256, 0, stream>>>(img, buf0, P(3), P(4), P(5), P(6), P(7), nullptr,
                                                    8, 512, 640, 512, 640, 1, 0);
        // c1: 8->8, k3
        conv_bn_relu_t<8, 3><<<g, 256, 0, stream>>>(buf0, buf1, P(8), P(9), P(10), P(11), P(12), nullptr,
                                                    8, 512, 640, 512, 640, 1, 0);
        // c2: 8->16, k5, s2 -> 256x320
        g = (16 * 256 * 320 + 255) / 256;
        conv_bn_relu_t<8, 5><<<g, 256, 0, stream>>>(buf1, buf0, P(13), P(14), P(15), P(16), P(17), nullptr,
                                                    16, 512, 640, 256, 320, 2, 0);
        // c3: 16->16, k3
        conv_bn_relu_t<16, 3><<<g, 256, 0, stream>>>(buf0, buf1, P(18), P(19), P(20), P(21), P(22), nullptr,
                                                     16, 256, 320, 256, 320, 1, 0);
        // c4: 16->16, k3
        conv_bn_relu_t<16, 3><<<g, 256, 0, stream>>>(buf1, buf0, P(23), P(24), P(25), P(26), P(27), nullptr,
                                                     16, 256, 320, 256, 320, 1, 0);
        // c5: 16->32, k5, s2 -> 128x160
        g = (32 * 128 * 160 + 255) / 256;
        conv_bn_relu_t<16, 5><<<g, 256, 0, stream>>>(buf0, buf1, P(28), P(29), P(30), P(31), P(32), nullptr,
                                                     32, 256, 320, 128, 160, 2, 0);
        // c6: 32->32, k3
        conv_bn_relu_t<32, 3><<<g, 256, 0, stream>>>(buf1, buf0, P(33), P(34), P(35), P(36), P(37), nullptr,
                                                     32, 128, 160, 128, 160, 1, 0);
        // feat: 32->32, k3, bias only
        conv_bn_relu_t<32, 3><<<g, 256, 0, stream>>>(buf0, fv, P(38), nullptr, nullptr, nullptr, nullptr,
                                                     P(39), 32, 128, 160, 128, 160, 1, 1);
    }
    const float* reff = feats;
    const float* src0 = feats + (size_t)1 * FC * NPIX;
    const float* src1 = feats + (size_t)2 * FC * NPIX;

    // ---- per-depth: warp+cost -> WMMA GRU chain -> prob ----
    const int wcGrid  = (NPIX * FC + 255) / 256;
    const int gruGrid = ((NPIX / 16) + 7) / 8;   // 8 waves/block, 16 px/wave
    const int pixGrid = (NPIX + 255) / 256;
    for (int d = 0; d < ND; ++d) {
        warp_cost_kernel<<<wcGrid, 256, 0, stream>>>(reff, src0, src1, Ms, dvals, d, xsl);
        gru_conv_wmma_t<32, 16, 9><<<gruGrid, 256, 0, stream>>>(xsl, g1u, g1c, P(43), P(45), h1);
        gru_conv_wmma_t<16, 4, 5><<<gruGrid, 256, 0, stream>>>(h1, g2u, g2c, P(49), P(51), h2);
        gru_conv_wmma_t<4, 2, 2><<<gruGrid, 256, 0, stream>>>(h2, g3u, g3c, P(55), P(57), h3);
        prob_conv_kernel<<<pixGrid, 256, 0, stream>>>(h3, P(58), P(59), regb, d);
    }

    // ---- softmax over depth -> output ----
    softmax_kernel<<<pixGrid, 256, 0, stream>>>(regb, (float*)d_out);
}
